// FullMatrix_Layer_50964081934338
// MI455X (gfx1250) — matrix operand order repaired and checked by reading, not compiled, not run on hardware
//
#include <hip/hip_runtime.h>
#include <hip/hip_bf16.h>

// ---------------------------------------------------------------------------
// Problem: D=512, N=2048.
//   q,k,v = h @ W^T ; scores[n,a,b] = q[n,a]*k[n,b]/sqrt(D); M=softmax_b;
//   o[n,a] = sum_b M[n,a,b]*v[n,b]; out = o @ W_O^T
// Key: softmax logits for fixed (n,a) are c*k[n,:] with scalar c=q[n,a]/sqrt(D),
// so the [N,D,D] tensor is never materialized.
// ---------------------------------------------------------------------------

#define N_TOK 2048
#define D_DIM 512

typedef __attribute__((ext_vector_type(16))) __bf16 bf16x16;
typedef __attribute__((ext_vector_type(8)))  __bf16 bf16x8;
typedef __attribute__((ext_vector_type(8)))  float  f32x8;

// ---------------------------------------------------------------------------
// fp32 -> bf16 convert (RNE via compiler __bf16 conversion)
// ---------------------------------------------------------------------------
__global__ __launch_bounds__(256) void f32_to_bf16_kernel(const float* __restrict__ src,
                                                          __bf16* __restrict__ dst, int n) {
    int i = blockIdx.x * 256 + threadIdx.x;
    if (i < n) dst[i] = (__bf16)src[i];
}

// ---------------------------------------------------------------------------
// C[MxD] = A[MxD](bf16) @ B[DxD](bf16, used transposed: C[m,j] = sum_k A[m,k]*B[j,k])
// One wave -> 32x32 output tile = 2x2 WMMA fragments, K loop in steps of 32.
// VGPR layouts per CDNA5 ISA 7.12.2 (wave32):
//   A 16x32 bf16 : lane m16 holds row m16; elems0-7 = K(8*hi ..), elems8-15 = K(16+8*hi ..)
//   B 32x16 bf16 : lane m16 holds out-col m16; elems0-7 = K(8*hi ..), elems8-15 = K(16+8*hi ..)
//   C/D 16x16 f32: elem r <-> row 8*hi + r, col m16
// ---------------------------------------------------------------------------
__global__ __launch_bounds__(128) void gemm_bt_wmma_kernel(const __bf16* __restrict__ A,
                                                           const __bf16* __restrict__ B,
                                                           float* __restrict__ C, int M) {
    const int lane = threadIdx.x & 31;
    const int wave = threadIdx.x >> 5;
    const int tilesPerRow = D_DIM / 32;                    // 16 column tiles
    const int tile = blockIdx.x * 4 + wave;
    const int tr = tile / tilesPerRow;
    const int tc = tile % tilesPerRow;
    const int row0 = tr * 32;
    const int col0 = tc * 32;
    if (row0 >= M) return;

    const int m16 = lane & 15;
    const int hi  = lane >> 4;

    f32x8 c00 = {}, c01 = {}, c10 = {}, c11 = {};

    const __bf16* arow0 = A + (size_t)(row0 + m16) * D_DIM;
    const __bf16* arow1 = arow0 + (size_t)16 * D_DIM;
    const __bf16* brow0 = B + (size_t)(col0 + m16) * D_DIM;
    const __bf16* brow1 = brow0 + (size_t)16 * D_DIM;

    for (int k = 0; k < D_DIM; k += 32) {
        // A fragments: two 16B chunks per lane, per row-tile
        bf16x8 a0lo = *(const bf16x8*)(arow0 + k + 8 * hi);
        bf16x8 a0hi = *(const bf16x8*)(arow0 + k + 16 + 8 * hi);
        bf16x8 a1lo = *(const bf16x8*)(arow1 + k + 8 * hi);
        bf16x8 a1hi = *(const bf16x8*)(arow1 + k + 16 + 8 * hi);
        bf16x16 a0 = __builtin_shufflevector(a0lo, a0hi, 0,1,2,3,4,5,6,7,8,9,10,11,12,13,14,15);
        bf16x16 a1 = __builtin_shufflevector(a1lo, a1hi, 0,1,2,3,4,5,6,7,8,9,10,11,12,13,14,15);
        // B fragments: two 16B chunks per lane, same K order as A
        bf16x8 b0lo = *(const bf16x8*)(brow0 + k + 8 * hi);
        bf16x8 b0hi = *(const bf16x8*)(brow0 + k + 16 + 8 * hi);
        bf16x8 b1lo = *(const bf16x8*)(brow1 + k + 8 * hi);
        bf16x8 b1hi = *(const bf16x8*)(brow1 + k + 16 + 8 * hi);
        bf16x16 b0 = __builtin_shufflevector(b0lo, b0hi, 0,1,2,3,4,5,6,7,8,9,10,11,12,13,14,15);
        bf16x16 b1 = __builtin_shufflevector(b1lo, b1hi, 0,1,2,3,4,5,6,7,8,9,10,11,12,13,14,15);

        // (neg_a, A, neg_b, B, c_mod, C, reuse_a, reuse_b)
        c00 = __builtin_amdgcn_wmma_f32_16x16x32_bf16(false, a0, false, b0, (short)0, c00, false, false);
        c01 = __builtin_amdgcn_wmma_f32_16x16x32_bf16(false, a0, false, b1, (short)0, c01, false, false);
        c10 = __builtin_amdgcn_wmma_f32_16x16x32_bf16(false, a1, false, b0, (short)0, c10, false, false);
        c11 = __builtin_amdgcn_wmma_f32_16x16x32_bf16(false, a1, false, b1, (short)0, c11, false, false);
    }

#pragma unroll
    for (int r = 0; r < 8; ++r) {
        const int mr = 8 * hi + r;
        C[(size_t)(row0 + mr) * D_DIM + col0 + m16]           = c00[r];
        C[(size_t)(row0 + mr) * D_DIM + col0 + 16 + m16]      = c01[r];
        C[(size_t)(row0 + 16 + mr) * D_DIM + col0 + m16]      = c10[r];
        C[(size_t)(row0 + 16 + mr) * D_DIM + col0 + 16 + m16] = c11[r];
    }
}

// ---------------------------------------------------------------------------
// Per-token softmax-weighted sum. Block = one token (256 threads, 8 waves).
// o[n,a] = sum_b exp2(c2*k_b - m2) * v_b / sum_b exp2(c2*k_b - m2),
//   c2 = q[n,a]*log2(e)/sqrt(D), m2 = c2>=0 ? c2*max(k) : c2*min(k).
// k,v staged in LDS; all threads read the same LDS word per iter (broadcast).
// Writes bf16 o for the final WMMA GEMM.
// ---------------------------------------------------------------------------
__global__ __launch_bounds__(256) void attn_softmax_kernel(const float* __restrict__ Q,
                                                           const float* __restrict__ K,
                                                           const float* __restrict__ V,
                                                           __bf16* __restrict__ O) {
    __shared__ float ks[D_DIM];
    __shared__ float vs[D_DIM];
    __shared__ float red[256];

    const int n   = blockIdx.x;
    const int tid = threadIdx.x;

    const float* kp = K + (size_t)n * D_DIM;
    const float* vp = V + (size_t)n * D_DIM;

    const float k0 = kp[tid];
    const float k1 = kp[tid + 256];
    ks[tid]       = k0;
    ks[tid + 256] = k1;
    vs[tid]       = vp[tid];
    vs[tid + 256] = vp[tid + 256];

    // block max of k
    red[tid] = fmaxf(k0, k1);
    __syncthreads();
    for (int s = 128; s > 0; s >>= 1) {
        if (tid < s) red[tid] = fmaxf(red[tid], red[tid + s]);
        __syncthreads();
    }
    const float kmax = red[0];
    __syncthreads();
    // block min of k
    red[tid] = fminf(k0, k1);
    __syncthreads();
    for (int s = 128; s > 0; s >>= 1) {
        if (tid < s) red[tid] = fminf(red[tid], red[tid + s]);
        __syncthreads();
    }
    const float kmin = red[0];
    __syncthreads();

    const float LOG2E      = 1.4426950408889634f;
    const float INV_SQRT_D = 0.04419417382415922f;   // 1/sqrt(512)
    const float* qp = Q + (size_t)n * D_DIM;

    const float c0 = qp[tid]       * (INV_SQRT_D * LOG2E);
    const float c1 = qp[tid + 256] * (INV_SQRT_D * LOG2E);
    const float m0 = (c0 >= 0.f) ? c0 * kmax : c0 * kmin;
    const float m1 = (c1 >= 0.f) ? c1 * kmax : c1 * kmin;

    float s0 = 0.f, s1 = 0.f, d0 = 0.f, d1 = 0.f;
#pragma unroll 4
    for (int b = 0; b < D_DIM; ++b) {
        const float kb = ks[b];
        const float vb = vs[b];
        const float e0 = __builtin_amdgcn_exp2f(fmaf(c0, kb, -m0));
        const float e1 = __builtin_amdgcn_exp2f(fmaf(c1, kb, -m1));
        s0 += e0;
        s1 += e1;
        d0 = fmaf(e0, vb, d0);
        d1 = fmaf(e1, vb, d1);
    }

    __bf16* op = O + (size_t)n * D_DIM;
    op[tid]       = (__bf16)(d0 / s0);
    op[tid + 256] = (__bf16)(d1 / s1);
}

// ---------------------------------------------------------------------------
// Host-side orchestration.
// Workspace layout (bytes):
//   [0      , 2 MiB )  h   as bf16        (2048*512*2)
//   [2 MiB  , 2.5MiB)  W_Q as bf16        (512*512*2)
//   [2.5MiB , 3 MiB )  W_K as bf16
//   [3 MiB  , 3.5MiB)  W_V as bf16
//   [3.5MiB , 4 MiB )  W_O as bf16
//   [4 MiB  , 8 MiB )  Q  f32
//   [8 MiB  , 12MiB )  K  f32
//   [12MiB  , 16MiB )  V  f32
//   [16MiB  , 18MiB )  o  bf16
// Total 18 MiB.
// ---------------------------------------------------------------------------
extern "C" void kernel_launch(void* const* d_in, const int* in_sizes, int n_in,
                              void* d_out, int out_size, void* d_ws, size_t ws_size,
                              hipStream_t stream) {
    (void)in_sizes; (void)n_in; (void)out_size; (void)ws_size;

    const float* h  = (const float*)d_in[0];
    const float* WQ = (const float*)d_in[1];
    const float* WK = (const float*)d_in[2];
    const float* WV = (const float*)d_in[3];
    const float* WO = (const float*)d_in[4];
    float* out = (float*)d_out;

    char* ws = (char*)d_ws;
    const size_t MB = 1024 * 1024;
    __bf16* hb  = (__bf16*)(ws + 0);
    __bf16* wqb = (__bf16*)(ws + 2 * MB);
    __bf16* wkb = (__bf16*)(ws + 2 * MB + 512 * 1024);
    __bf16* wvb = (__bf16*)(ws + 3 * MB);
    __bf16* wob = (__bf16*)(ws + 3 * MB + 512 * 1024);
    float*  Qf  = (float*)(ws + 4 * MB);
    float*  Kf  = (float*)(ws + 8 * MB);
    float*  Vf  = (float*)(ws + 12 * MB);
    __bf16* Ob  = (__bf16*)(ws + 16 * MB);

    const int ND = N_TOK * D_DIM;   // 1048576
    const int DD = D_DIM * D_DIM;   // 262144

    // Stage 0: convert inputs to bf16
    f32_to_bf16_kernel<<<(ND + 255) / 256, 256, 0, stream>>>(h,  hb,  ND);
    f32_to_bf16_kernel<<<(DD + 255) / 256, 256, 0, stream>>>(WQ, wqb, DD);
    f32_to_bf16_kernel<<<(DD + 255) / 256, 256, 0, stream>>>(WK, wkb, DD);
    f32_to_bf16_kernel<<<(DD + 255) / 256, 256, 0, stream>>>(WV, wvb, DD);
    f32_to_bf16_kernel<<<(DD + 255) / 256, 256, 0, stream>>>(WO, wob, DD);

    // Stage 1: Q, K, V projections (WMMA bf16, f32 accumulate)
    const int waveTiles = (N_TOK / 32) * (D_DIM / 32);   // 64*16 = 1024
    const int gemmBlocks = waveTiles / 4;                // 4 waves per block
    gemm_bt_wmma_kernel<<<gemmBlocks, 128, 0, stream>>>(hb, wqb, Qf, N_TOK);
    gemm_bt_wmma_kernel<<<gemmBlocks, 128, 0, stream>>>(hb, wkb, Kf, N_TOK);
    gemm_bt_wmma_kernel<<<gemmBlocks, 128, 0, stream>>>(hb, wvb, Vf, N_TOK);

    // Stage 2: per-token rank-1 softmax attention
    attn_softmax_kernel<<<N_TOK, 256, 0, stream>>>(Qf, Kf, Vf, Ob);

    // Stage 3: output projection into d_out
    gemm_bt_wmma_kernel<<<gemmBlocks, 128, 0, stream>>>(Ob, wob, out, N_TOK);
}
